// Decoder_82351702934307
// MI455X (gfx1250) — compile-verified, matrix operand order checked by reading, not run on hardware
//
#include <hip/hip_runtime.h>
#include <hip/hip_bf16.h>

// Problem constants (fixed by the reference setup).
#define B_   1024   // batch
#define H_   1024   // hidden
#define S_   128    // steps
#define E_   8      // exogenous features
#define G4_  4096   // 4*H (gates i,f,g,o stacked)

typedef __bf16 v16bf __attribute__((ext_vector_type(16)));
typedef float  v8f   __attribute__((ext_vector_type(8)));

union FragBF {
    v16bf v;
    uint4 q[2];   // two 16-byte halves (8 bf16 each)
};

__device__ __forceinline__ unsigned short f2bf_u(float f) {
    union { float f; unsigned u; } v; v.f = f;
    unsigned r = v.u + 0x7FFFu + ((v.u >> 16) & 1u);  // round-to-nearest-even
    return (unsigned short)(r >> 16);
}

__device__ __forceinline__ float sigf(float x)      { return 1.0f / (1.0f + __expf(-x)); }
__device__ __forceinline__ float tanh_fast(float x) { return 1.0f - 2.0f / (1.0f + __expf(2.0f * x)); }

// ---------------------------------------------------------------------------
// Fused GEMM (WMMA bf16 -> f32 accum) + LSTM cell.
//   z = A @ W^T (+ bias) (+ layer0: rank-9 [prev_y, exo] @ Wih0^T)
//   c' = sig(f)*c + sig(i)*tanh(g);  h' = sig(o)*tanh(c')
//
// Workgroup tile: 128 batch rows x 64 hidden cols x 4 gates.
// 512 threads = 16 waves laid out 4(M) x 4(N); per wave: 2 M-subtiles x 4
// gates = 8 v8f accumulators. K is streamed in 64-wide chunks through
// double-buffered LDS (one barrier per 16 WMMAs).
// ---------------------------------------------------------------------------
template<int LAYER>
__global__ __launch_bounds__(512)
void lstm_layer_kernel(const unsigned short* __restrict__ Alo,   // bf16 [B,1024] (L0: h0 ; L1: h0n)
                       const unsigned short* __restrict__ Ahi,   // bf16 [B,1024] (L1: h1 prev)
                       const unsigned short* __restrict__ Wbf,   // bf16 [4096, KT]
                       const float* __restrict__ bias,           // [4096] combined
                       float* __restrict__ cstate,               // f32 [B,1024], in-place
                       unsigned short* __restrict__ Hbf_out,     // bf16 [B,1024]
                       float* __restrict__ Hf32_out,             // f32 [B,1024] (L1 only)
                       const float* __restrict__ prev_y,         // [B]   (L0 only)
                       const float* __restrict__ exo,            // &x_future[t*E], row stride S*E (L0)
                       const float* __restrict__ Wih9)           // [4096,9] (L0 only)
{
    constexpr int KT = (LAYER == 0) ? 1024 : 2048;
    constexpr int KC = 64;                      // K chunk

    // double-buffered staging: A 128x64, B 256x64 (bf16)
    __shared__ __align__(16) unsigned short ldsA[2][128 * KC];   // 2 x 16 KB
    __shared__ __align__(16) unsigned short ldsB[2][256 * KC];   // 2 x 32 KB

    const int tid    = threadIdx.x;
    const int lane   = tid & 31;
    const int wave   = tid >> 5;          // 0..15
    const int wm     = wave >> 2;         // 0..3 : 32-row M sub-block
    const int wn     = wave & 3;          // 0..3 : 16-col N subtile
    const int mblock = blockIdx.y * 128;
    const int nblock = blockIdx.x * 64;

    v8f acc[8];                           // [s*4 + gate]
    #pragma unroll
    for (int i = 0; i < 8; ++i)
        #pragma unroll
        for (int e = 0; e < 8; ++e) acc[i][e] = 0.0f;

    // cooperative loader mappings (512 threads)
    // A: 128 rows x 64 k -> 16 elems (2 x uint4) per thread
    const int arow = tid >> 2;            // 0..127
    const int akc  = (tid & 3) * 16;      // 0,16,32,48 (bf16 elems)
    // B: 256 rows (4 gates x 64 cols) x 64 k -> 32 elems (4 x uint4) per thread
    const int brr  = tid >> 1;            // 0..255
    const int bkc  = (tid & 1) * 32;      // 0,32
    const int bgate = brr >> 6;           // 0..3
    const int bn    = brr & 63;           // 0..63
    const long brow = (long)(bgate * H_ + nblock + bn) * KT;

    // WMMA 16-bit fragment lane layout: lanes 0-15 -> K {0..7,16..23},
    // lanes 16-31 -> K {8..15,24..31}
    const int halfsel = lane >> 4;
    const int kb      = halfsel * 8;
    const int l15     = lane & 15;

    // ---- stage one K-chunk into the selected LDS buffer ----
    auto stage = [&](int kc, int bufsel) {
        const unsigned short* asrc = Alo;
        int acol = kc;
        if (LAYER == 1 && kc >= 1024) { asrc = Ahi; acol = kc - 1024; }
        const unsigned short* agp = &asrc[(long)(mblock + arow) * H_ + acol + akc];
        uint4 a0 = *(const uint4*)(agp + 0);
        uint4 a1 = *(const uint4*)(agp + 8);
        const unsigned short* bgp = &Wbf[brow + kc + bkc];
        uint4 w0 = *(const uint4*)(bgp + 0);
        uint4 w1 = *(const uint4*)(bgp + 8);
        uint4 w2 = *(const uint4*)(bgp + 16);
        uint4 w3 = *(const uint4*)(bgp + 24);
        // prefetch the weight stream one chunk ahead, WGP scope (high
        // locality) so L2-resident lines land in the WGP cache.
        if (kc + 2 * KC <= KT)
            __builtin_prefetch(&Wbf[brow + kc + KC + bkc], 0, 3);
        *(uint4*)&ldsA[bufsel][arow * KC + akc + 0] = a0;
        *(uint4*)&ldsA[bufsel][arow * KC + akc + 8] = a1;
        unsigned short* bq = &ldsB[bufsel][brr * KC + bkc];
        *(uint4*)(bq + 0)  = w0;
        *(uint4*)(bq + 8)  = w1;
        *(uint4*)(bq + 16) = w2;
        *(uint4*)(bq + 24) = w3;
    };

    // prologue: stage chunk 0 into buffer 0
    stage(0, 0);
    __syncthreads();

    for (int kc = 0; kc < KT; kc += KC) {
        const int cur = (kc >> 6) & 1;
        if (kc + KC < KT) stage(kc + KC, cur ^ 1);   // overlaps with WMMAs below

        // ---- compute from lds[cur]: 2 k-steps of 32 ----
        #pragma unroll
        for (int ks = 0; ks < 2; ++ks) {
            const int ko = ks * 32;
            FragBF fa[2], fb[4];
            #pragma unroll
            for (int s = 0; s < 2; ++s) {
                const int row = wm * 32 + s * 16 + l15;
                fa[s].q[0] = *(const uint4*)&ldsA[cur][row * KC + ko + kb];
                fa[s].q[1] = *(const uint4*)&ldsA[cur][row * KC + ko + kb + 16];
            }
            #pragma unroll
            for (int g = 0; g < 4; ++g) {
                const int row = g * 64 + wn * 16 + l15;
                fb[g].q[0] = *(const uint4*)&ldsB[cur][row * KC + ko + kb];
                fb[g].q[1] = *(const uint4*)&ldsB[cur][row * KC + ko + kb + 16];
            }
            #pragma unroll
            for (int s = 0; s < 2; ++s)
                #pragma unroll
                for (int g = 0; g < 4; ++g)
                    acc[s * 4 + g] = __builtin_amdgcn_wmma_f32_16x16x32_bf16(
                        false, fa[s].v, false, fb[g].v,
                        (short)0, acc[s * 4 + g], false, false);
        }
        // one barrier per chunk: reads of lds[cur] done + writes of lds[cur^1]
        // visible before next iteration flips the buffers.
        __syncthreads();
    }

    // -------- epilogue: bias, (layer0) rank-9 input, LSTM cell --------
    const int j = nblock + wn * 16 + l15;          // hidden column
    const float bi = bias[0 * H_ + j];
    const float bf = bias[1 * H_ + j];
    const float bg = bias[2 * H_ + j];
    const float bo = bias[3 * H_ + j];

    float w9[4][9];
    if (LAYER == 0) {
        #pragma unroll
        for (int g = 0; g < 4; ++g)
            #pragma unroll
            for (int e = 0; e < 9; ++e)
                w9[g][e] = Wih9[(long)(g * H_ + j) * 9 + e];
    }

    #pragma unroll
    for (int s = 0; s < 2; ++s) {
        #pragma unroll
        for (int r = 0; r < 8; ++r) {
            // C/D layout: VGPR r -> M=r (lanes 0-15) or M=r+8 (lanes 16-31)
            const int m = mblock + wm * 32 + s * 16 + r + halfsel * 8;
            float zi = acc[s * 4 + 0][r] + bi;
            float zf = acc[s * 4 + 1][r] + bf;
            float zg = acc[s * 4 + 2][r] + bg;
            float zo = acc[s * 4 + 3][r] + bo;
            if (LAYER == 0) {
                float xin[9];
                xin[0] = prev_y[m];
                #pragma unroll
                for (int e = 0; e < 8; ++e)
                    xin[1 + e] = exo[(long)m * (S_ * E_) + e];
                #pragma unroll
                for (int e = 0; e < 9; ++e) {
                    zi += xin[e] * w9[0][e];
                    zf += xin[e] * w9[1][e];
                    zg += xin[e] * w9[2][e];
                    zo += xin[e] * w9[3][e];
                }
            }
            const long idx = (long)m * H_ + j;
            const float cold = cstate[idx];
            const float cn = sigf(zf) * cold + sigf(zi) * tanh_fast(zg);
            const float hn = sigf(zo) * tanh_fast(cn);
            cstate[idx]  = cn;
            Hbf_out[idx] = f2bf_u(hn);
            if (LAYER == 1) Hf32_out[idx] = hn;
        }
    }
}

// ---------------------------------------------------------------------------
// yhat[b] = h1[b,:] . Wp[0,:] + bp ; one wave per batch row, shuffle-reduce.
// Also forwards prev_y for the next step's layer-0 input.
// ---------------------------------------------------------------------------
__global__ __launch_bounds__(256)
void proj_kernel(const float* __restrict__ h1f, const float* __restrict__ Wp,
                 const float* __restrict__ bp, float* __restrict__ out,
                 float* __restrict__ prev_y, int t)
{
    const int row  = blockIdx.x * 8 + (threadIdx.x >> 5);
    const int lane = threadIdx.x & 31;
    float s = 0.0f;
    #pragma unroll 8
    for (int i = 0; i < 32; ++i) {
        const int jj = i * 32 + lane;
        s += h1f[(long)row * H_ + jj] * Wp[jj];
    }
    #pragma unroll
    for (int off = 16; off > 0; off >>= 1)
        s += __shfl_xor(s, off, 32);
    if (lane == 0) {
        const float y = s + bp[0];
        out[(long)row * S_ + t] = y;
        prev_y[row] = y;
    }
}

// ---------------------------------------------------------------------------
// One-time setup kernels
// ---------------------------------------------------------------------------
__global__ __launch_bounds__(256)
void cvt_w0_kernel(const float* __restrict__ src, unsigned short* __restrict__ dst)
{
    const long i = (long)blockIdx.x * 256 + threadIdx.x;   // over 4096*1024
    dst[i] = f2bf_u(src[i]);
}

__global__ __launch_bounds__(256)
void build_wcat_kernel(const float* __restrict__ wih1, const float* __restrict__ whh1,
                       unsigned short* __restrict__ dst)
{
    const long i = (long)blockIdx.x * 256 + threadIdx.x;   // over 4096*2048
    const int n = (int)(i >> 11);
    const int k = (int)(i & 2047);
    const float v = (k < 1024) ? wih1[(long)n * 1024 + k]
                               : whh1[(long)n * 1024 + (k - 1024)];
    dst[i] = f2bf_u(v);
}

__global__ __launch_bounds__(256)
void bias_kernel(const float* __restrict__ bih0, const float* __restrict__ bhh0,
                 const float* __restrict__ bih1, const float* __restrict__ bhh1,
                 float* __restrict__ b0, float* __restrict__ b1)
{
    const int i = blockIdx.x * 256 + threadIdx.x;          // over 4096
    b0[i] = bih0[i] + bhh0[i];
    b1[i] = bih1[i] + bhh1[i];
}

__global__ __launch_bounds__(256)
void init_state_kernel(const float* __restrict__ h0, const float* __restrict__ c0in,
                       const float* __restrict__ y0,
                       float* __restrict__ c0s, float* __restrict__ c1s,
                       unsigned short* __restrict__ h0bf, unsigned short* __restrict__ h1bf,
                       float* __restrict__ py)
{
    const long i = (long)blockIdx.x * 256 + threadIdx.x;   // over B*H
    c0s[i] = c0in[i];
    c1s[i] = c0in[(long)B_ * H_ + i];
    h0bf[i] = f2bf_u(h0[i]);
    h1bf[i] = f2bf_u(h0[(long)B_ * H_ + i]);
    if (i < B_) py[i] = y0[i];
}

// ---------------------------------------------------------------------------
extern "C" void kernel_launch(void* const* d_in, const int* in_sizes, int n_in,
                              void* d_out, int out_size, void* d_ws, size_t ws_size,
                              hipStream_t stream)
{
    const float* y0   = (const float*)d_in[0];
    const float* xfut = (const float*)d_in[1];
    const float* h0in = (const float*)d_in[2];
    const float* c0in = (const float*)d_in[3];
    const float* Wih0 = (const float*)d_in[4];
    const float* Whh0 = (const float*)d_in[5];
    const float* bih0 = (const float*)d_in[6];
    const float* bhh0 = (const float*)d_in[7];
    const float* Wih1 = (const float*)d_in[8];
    const float* Whh1 = (const float*)d_in[9];
    const float* bih1 = (const float*)d_in[10];
    const float* bhh1 = (const float*)d_in[11];
    const float* Wp   = (const float*)d_in[12];
    const float* bp   = (const float*)d_in[13];
    (void)in_sizes; (void)n_in; (void)out_size; (void)ws_size;

    // carve workspace (256B aligned slabs); total ~44 MB
    char* ws = (char*)d_ws;
    size_t off = 0;
    auto carve = [&](size_t bytes) -> void* {
        void* p = ws + off;
        off = (off + bytes + 255) & ~(size_t)255;
        return p;
    };
    unsigned short* W0bf   = (unsigned short*)carve((size_t)G4_ * H_   * 2);  // 8 MB
    unsigned short* Wc1bf  = (unsigned short*)carve((size_t)G4_ * 2048 * 2);  // 16 MB
    float* b0              = (float*)carve((size_t)G4_ * 4);
    float* b1              = (float*)carve((size_t)G4_ * 4);
    float* c0s             = (float*)carve((size_t)B_ * H_ * 4);              // 4 MB
    float* c1s             = (float*)carve((size_t)B_ * H_ * 4);              // 4 MB
    unsigned short* H0bf[2];
    unsigned short* H1bf[2];
    H0bf[0] = (unsigned short*)carve((size_t)B_ * H_ * 2);                    // 2 MB each
    H0bf[1] = (unsigned short*)carve((size_t)B_ * H_ * 2);
    H1bf[0] = (unsigned short*)carve((size_t)B_ * H_ * 2);
    H1bf[1] = (unsigned short*)carve((size_t)B_ * H_ * 2);
    float* h1f             = (float*)carve((size_t)B_ * H_ * 4);              // 4 MB
    float* py              = (float*)carve((size_t)B_ * 4);

    const dim3 blk(256);

    // ---- setup ----
    cvt_w0_kernel<<<(G4_ * H_) / 256, blk, 0, stream>>>(Whh0, W0bf);
    build_wcat_kernel<<<(G4_ * 2048) / 256, blk, 0, stream>>>(Wih1, Whh1, Wc1bf);
    bias_kernel<<<G4_ / 256, blk, 0, stream>>>(bih0, bhh0, bih1, bhh1, b0, b1);
    init_state_kernel<<<(B_ * H_) / 256, blk, 0, stream>>>(h0in, c0in, y0, c0s, c1s,
                                                           H0bf[0], H1bf[0], py);

    // ---- sequential decode: 3 kernels per step ----
    const dim3 ggrid(16, 8);    // 16 N-blocks x 8 M-blocks (128-row M tiles)
    const dim3 gblk(512);
    float* out = (float*)d_out; // [B, S]
    for (int t = 0; t < S_; ++t) {
        const int pp = t & 1;
        // layer 0: h0' from h0 (ping) + [prev_y, exo_t]; writes h0 (pong)
        lstm_layer_kernel<0><<<ggrid, gblk, 0, stream>>>(
            H0bf[pp], nullptr, W0bf, b0, c0s,
            H0bf[pp ^ 1], nullptr, py, xfut + (long)t * E_, Wih0);
        // layer 1: h1' from [h0'(pong) | h1(ping)] vs Wcat; writes h1 (pong) + f32
        lstm_layer_kernel<1><<<ggrid, gblk, 0, stream>>>(
            H0bf[pp ^ 1], H1bf[pp], Wc1bf, b1, c1s,
            H1bf[pp ^ 1], h1f, nullptr, nullptr, nullptr);
        // projection + prev_y forward
        proj_kernel<<<B_ / 8, blk, 0, stream>>>(h1f, Wp, bp, out, py, t);
    }
}
